// EnhancedVLM_54606214201762
// MI455X (gfx1250) — matrix operand order repaired and checked by reading, not compiled, not run on hardware
//
#include <hip/hip_runtime.h>
#include <hip/hip_bf16.h>

// ---------------------------------------------------------------------------
// EnhancedVLM SSM stack for MI455X (gfx1250, wave32, WMMA + TDM)
//   L=4, S=64, H=256, D_IN=D_OUT=768, B=8, T=2048 -> NT=16384 tokens
// Dense GEMMs in bf16 WMMA (f32 accum). The block's 16xK activation tile is
// DMA'd into LDS via the Tensor Data Mover and shared by all 4 waves; weight
// fragments stream from global (L2-resident). Serial recurrence restructured
// so the 64x256 C-projection is lifted out of the scan into a bulk WMMA GEMM.
// ---------------------------------------------------------------------------

#define NT      16384      // B*T tokens
#define L_LAY   4
#define S_DIM   64
#define H_DIM   256
#define D_INOUT 768
#define B_SZ    8
#define T_SEQ   2048
#define K_MAX   768

typedef __attribute__((ext_vector_type(16))) __bf16        v16bf;
typedef __attribute__((ext_vector_type(8)))  __bf16        v8bf;
typedef __attribute__((ext_vector_type(8)))  float         v8f;
typedef __attribute__((ext_vector_type(4)))  unsigned int  v4u;
typedef __attribute__((ext_vector_type(8)))  int           v8i;
typedef __attribute__((ext_vector_type(4)))  int           v4i;

#define HAVE_TDM (__has_builtin(__builtin_amdgcn_tensor_load_to_lds) && \
                  __has_builtin(__builtin_amdgcn_s_wait_tensorcnt))

static __device__ __forceinline__ unsigned short f32_to_bf16(float f) {
    unsigned int u = __float_as_uint(f);
    u += 0x7FFFu + ((u >> 16) & 1u);            // round-to-nearest-even
    return (unsigned short)(u >> 16);
}

// ------------------------------- fp32 -> bf16 ------------------------------
__global__ void cvt_bf16_kernel(const float* __restrict__ src,
                                unsigned short* __restrict__ dst, int n) {
    int i = blockIdx.x * 256 + threadIdx.x;
    if (i < n) dst[i] = f32_to_bf16(src[i]);
}

// ------------------------------- WMMA GEMM ---------------------------------
// C[M,N] = act( A[M,K] (bf16, row-major) * W[N,K]^T (bf16, row-major) + bias[N] )
// block = 128 threads = 4 waves, each wave owns one 16x16 output tile.
// grid  = (M/16, N/64). A-tile (16xK) staged in LDS via TDM, shared by waves.
template <int ACT, bool STORE_BF16>
__global__ __launch_bounds__(128)
void wmma_gemm_kernel(const unsigned short* __restrict__ A,
                      const unsigned short* __restrict__ W,
                      const float* __restrict__ bias,
                      float* __restrict__ C,
                      unsigned short* __restrict__ C2,
                      int M, int N, int K) {
    __shared__ __align__(16) unsigned short Atile[16 * K_MAX];   // 24 KB max

    const int wave = threadIdx.x >> 5;
    const int lane = threadIdx.x & 31;
    const int r    = lane & 15;            // row/col within tile
    const int hi   = lane >> 4;            // half-wave selector

    const int mtile = blockIdx.x;
    const int ntile = blockIdx.y * 4 + wave;

    const unsigned short* __restrict__ Ag = A + (size_t)mtile * 16 * K;

#if HAVE_TDM
    // ---- Tensor Data Mover: 2D tile (16 rows x K bf16) global -> LDS ----
    // D# per CDNA5 ISA. Issued once, by wave 0 only.
    // 6-arg builtin variant: (v4u g0, v8i g1, v4i g2, v4i g3, v8i g4, i32 cpol)
    if (threadIdx.x < 32) {
        const unsigned       lds_off = (unsigned)(uintptr_t)&Atile[0];
        const unsigned long long ga  = (unsigned long long)(uintptr_t)Ag;
        v4u g0;
        g0[0] = 1u;                                    // count=1, valid user D#
        g0[1] = lds_off;                               // lds_addr
        g0[2] = (unsigned)ga;                          // global_addr[31:0]
        g0[3] = (unsigned)((ga >> 32) & 0x1FFFFFFull)  // global_addr[56:32]
              | (2u << 30);                            // type=2 ("image")
        v8i g1;
        g1[0] = (int)(1u << 16);                       // data_size=1 (2 bytes)
        g1[1] = (int)(((unsigned)K & 0xFFFFu) << 16);  // tensor_dim0[15:0]
        g1[2] = (int)(((unsigned)K >> 16) | (16u << 16)); // dim0 hi | tensor_dim1=16
        g1[3] = (int)(((unsigned)K & 0xFFFFu) << 16);  // tile_dim0 = K
        g1[4] = 16;                                    // tile_dim1=16, tile_dim2=0
        g1[5] = K;                                     // tensor_dim0_stride lo
        g1[6] = 0;
        g1[7] = 0;
        v4i gz4; gz4[0] = 0; gz4[1] = 0; gz4[2] = 0; gz4[3] = 0;
        v8i gz8; gz8[0] = 0; gz8[1] = 0; gz8[2] = 0; gz8[3] = 0;
                 gz8[4] = 0; gz8[5] = 0; gz8[6] = 0; gz8[7] = 0;
        __builtin_amdgcn_tensor_load_to_lds(g0, g1, gz4, gz4, gz8, 0);
        __builtin_amdgcn_s_wait_tensorcnt(0);
    }
#else
    // fallback: cooperative 16-byte copy
    for (int i = (int)threadIdx.x; i < 2 * K; i += 128)
        ((float4*)Atile)[i] = ((const float4*)Ag)[i];
#endif
    __syncthreads();

    const unsigned short* Arow = Atile + r * K;                  // LDS
    const unsigned short* Wrow = W + (size_t)(ntile * 16 + r) * K;

    v8f acc = {};
    for (int k0 = 0; k0 < K; k0 += 32) {
        // A fragment (16x32 bf16, ISA 7.12.2): lanes 0-15 hold K{0..7,16..23},
        // lanes 16-31 hold K{8..15,24..31} of row r.  (ds_load_b128 x2)
        v8bf alo = *(const v8bf*)(Arow + k0 + 8 * hi);
        v8bf ahi = *(const v8bf*)(Arow + k0 + 16 + 8 * hi);
        // B fragment (32x16 bf16): lanes 0-15 hold K{0..7,16..23}, lanes 16-31 K{8..15,24..31}
        // of column (ntile*16+r) == row of W.          (global_load_b128 x2)
        v8bf blo = *(const v8bf*)(Wrow + k0 + 8 * hi);
        v8bf bhi = *(const v8bf*)(Wrow + k0 + 16 + 8 * hi);
        v16bf av = __builtin_shufflevector(alo, ahi, 0,1,2,3,4,5,6,7,8,9,10,11,12,13,14,15);
        v16bf bv = __builtin_shufflevector(blo, bhi, 0,1,2,3,4,5,6,7,8,9,10,11,12,13,14,15);
        acc = __builtin_amdgcn_wmma_f32_16x16x32_bf16(
            /*neg_a=*/false, av, /*neg_b=*/false, bv,
            /*c_mod=*/(short)0, acc, /*reuse_a=*/false, /*reuse_b=*/false);
    }

    const int   col = ntile * 16 + r;
    const float bv  = bias ? bias[col] : 0.0f;

#pragma unroll
    for (int i = 0; i < 8; ++i) {
        float v = acc[i] + bv;
        if (ACT == 1) v = 1.0f / (1.0f + __expf(-v));   // sigmoid (gate)
        const size_t idx = (size_t)(mtile * 16 + i + 8 * hi) * N + col;
        C[idx] = v;
        if (STORE_BF16) C2[idx] = f32_to_bf16(v);
    }
}

// ------------------------------- LayerNorm ---------------------------------
// one 256-thread block per token, LDS tree reduction over H=256
__global__ void ln_kernel(const float* __restrict__ h,
                          const float* __restrict__ g,
                          const float* __restrict__ b,
                          float* __restrict__ xn,
                          unsigned short* __restrict__ xnb) {
    __shared__ float red[H_DIM];
    const int tok = blockIdx.x, tid = threadIdx.x;
    const size_t idx = (size_t)tok * H_DIM + tid;
    const float v = h[idx];

    red[tid] = v; __syncthreads();
    for (int s = H_DIM / 2; s > 0; s >>= 1) {
        if (tid < s) red[tid] += red[tid + s];
        __syncthreads();
    }
    const float mu = red[0] * (1.0f / H_DIM);
    __syncthreads();

    const float d = v - mu;
    red[tid] = d * d; __syncthreads();
    for (int s = H_DIM / 2; s > 0; s >>= 1) {
        if (tid < s) red[tid] += red[tid + s];
        __syncthreads();
    }
    const float var = red[0] * (1.0f / H_DIM);

    const float o = d * rsqrtf(var + 1e-5f) * g[tid] + b[tid];
    xn[idx]  = o;
    xnb[idx] = f32_to_bf16(o);
}

// ------------------------------- serial scan -------------------------------
// grid = B (one block per batch), block = 256 threads. A kept in LDS.
// h_new[s] computed by 4 lanes (16 FMAs each) + 2 xor-shuffles; states stored
// as bf16 so that y = states @ Cm^T runs as a bulk WMMA GEMM afterwards.
__global__ void scan_kernel(const float* __restrict__ xs,      // [B,T,S]
                            const float* __restrict__ A,       // [S,S]
                            const float* __restrict__ Bv,      // [S]
                            const float* __restrict__ biasA,   // [S]
                            unsigned short* __restrict__ states /* [B,T,S] bf16 */) {
    __shared__ float A_s[S_DIM * S_DIM];
    __shared__ float h_s[S_DIM];
    __shared__ float Bv_s[S_DIM];
    __shared__ float bA_s[S_DIM];

    const int tid = threadIdx.x;
    for (int i = tid; i < S_DIM * S_DIM; i += 256) A_s[i] = A[i];
    if (tid < S_DIM) { Bv_s[tid] = Bv[tid]; bA_s[tid] = biasA[tid]; h_s[tid] = 0.0f; }
    __syncthreads();

    const int s    = tid >> 2;
    const int part = tid & 3;
    const int k0   = part * 16;
    const float* __restrict__ Ar = &A_s[s * S_DIM + k0];
    const size_t base = (size_t)blockIdx.x * T_SEQ * S_DIM;

    for (int t = 0; t < T_SEQ; ++t) {
        const float xv = xs[base + (size_t)t * S_DIM + s];
        if (t + 1 < T_SEQ)
            __builtin_prefetch(xs + base + (size_t)(t + 1) * S_DIM + s, 0, 0);

        float acc = 0.0f;
#pragma unroll
        for (int j = 0; j < 16; ++j) acc = fmaf(Ar[j], h_s[k0 + j], acc);
        acc += __shfl_xor(acc, 1, 32);
        acc += __shfl_xor(acc, 2, 32);

        __syncthreads();                       // all reads of h_s done
        if (part == 0) {
            float hv = acc + Bv_s[s] * xv + bA_s[s];
            hv = fminf(10.0f, fmaxf(-10.0f, hv));
            h_s[s] = hv;
            states[base + (size_t)t * S_DIM + s] = f32_to_bf16(hv);
        }
        __syncthreads();                       // new h visible
    }
}

// --------------------------- gated residual combine ------------------------
// h = h + gate*yproj + (1-gate)*xn ; also emit bf16 copy of h for next GEMM
__global__ void combine_kernel(float* __restrict__ h,
                               unsigned short* __restrict__ hb,
                               const float* __restrict__ xn,
                               const float* __restrict__ gate,
                               const float* __restrict__ yp, int n) {
    int i = blockIdx.x * 256 + threadIdx.x;
    if (i >= n) return;
    const float g = gate[i];
    const float v = h[i] + g * yp[i] + (1.0f - g) * xn[i];
    h[i]  = v;
    hb[i] = f32_to_bf16(v);
}

// ---------------------------------------------------------------------------
static void launch_gemm(const unsigned short* A, const unsigned short* W,
                        const float* bias, float* C, unsigned short* C2,
                        int M, int N, int K, int act, hipStream_t st) {
    dim3 grid(M / 16, N / 64), block(128);
    if (act == 1)
        wmma_gemm_kernel<1, false><<<grid, block, 0, st>>>(A, W, bias, C, nullptr, M, N, K);
    else if (C2)
        wmma_gemm_kernel<0, true><<<grid, block, 0, st>>>(A, W, bias, C, C2, M, N, K);
    else
        wmma_gemm_kernel<0, false><<<grid, block, 0, st>>>(A, W, bias, C, nullptr, M, N, K);
}

static void launch_cvt(const float* s, unsigned short* d, int n, hipStream_t st) {
    cvt_bf16_kernel<<<(n + 255) / 256, 256, 0, st>>>(s, d, n);
}

extern "C" void kernel_launch(void* const* d_in, const int* in_sizes, int n_in,
                              void* d_out, int out_size, void* d_ws, size_t ws_size,
                              hipStream_t stream) {
    (void)in_sizes; (void)n_in; (void)out_size; (void)ws_size;

    const float* x          = (const float*)d_in[0];
    const float* in_proj_w  = (const float*)d_in[1];
    const float* in_proj_b  = (const float*)d_in[2];
    const float* ln_g       = (const float*)d_in[3];
    const float* ln_b       = (const float*)d_in[4];
    const float* ip_w       = (const float*)d_in[5];
    const float* ip_b       = (const float*)d_in[6];
    const float* A_mat      = (const float*)d_in[7];
    const float* Bv_vec     = (const float*)d_in[8];
    const float* Cm         = (const float*)d_in[9];
    const float* bias_A     = (const float*)d_in[10];
    const float* bias_C     = (const float*)d_in[11];
    const float* gate_w     = (const float*)d_in[12];
    const float* gate_b     = (const float*)d_in[13];
    const float* proj_w     = (const float*)d_in[14];
    const float* proj_b     = (const float*)d_in[15];
    const float* out_proj_w = (const float*)d_in[16];
    const float* out_proj_b = (const float*)d_in[17];

    // ---------------- workspace carve-up (256B aligned) ----------------
    char* ws = (char*)d_ws;
    size_t off = 0;
    auto carve = [&](size_t bytes) -> char* {
        char* p = ws + off;
        off += (bytes + 255) & ~(size_t)255;
        return p;
    };
    unsigned short* xb     = (unsigned short*)carve((size_t)NT * D_INOUT * 2);
    unsigned short* wib    = (unsigned short*)carve((size_t)H_DIM * D_INOUT * 2);
    unsigned short* wip    = (unsigned short*)carve((size_t)L_LAY * S_DIM * H_DIM * 2);
    unsigned short* wgate  = (unsigned short*)carve((size_t)L_LAY * H_DIM * H_DIM * 2);
    unsigned short* wproj  = (unsigned short*)carve((size_t)L_LAY * H_DIM * H_DIM * 2);
    unsigned short* wcm    = (unsigned short*)carve((size_t)L_LAY * H_DIM * S_DIM * 2);
    unsigned short* wout   = (unsigned short*)carve((size_t)D_INOUT * H_DIM * 2);
    float*          hbuf   = (float*)carve((size_t)NT * H_DIM * 4);
    unsigned short* hb     = (unsigned short*)carve((size_t)NT * H_DIM * 2);
    float*          xnbuf  = (float*)carve((size_t)NT * H_DIM * 4);
    unsigned short* xnb    = (unsigned short*)carve((size_t)NT * H_DIM * 2);
    float*          xstate = (float*)carve((size_t)NT * S_DIM * 4);
    float*          gatebf = (float*)carve((size_t)NT * H_DIM * 4);
    unsigned short* stb    = (unsigned short*)carve((size_t)NT * S_DIM * 2);
    float*          ybuf   = (float*)carve((size_t)NT * H_DIM * 4);
    unsigned short* yb     = (unsigned short*)carve((size_t)NT * H_DIM * 2);
    float*          yproj  = (float*)carve((size_t)NT * H_DIM * 4);

    // ---------------- one-time conversions to bf16 ----------------
    launch_cvt(x,          xb,    NT * D_INOUT,               stream);
    launch_cvt(in_proj_w,  wib,   H_DIM * D_INOUT,            stream);
    launch_cvt(ip_w,       wip,   L_LAY * S_DIM * H_DIM,      stream);
    launch_cvt(gate_w,     wgate, L_LAY * H_DIM * H_DIM,      stream);
    launch_cvt(proj_w,     wproj, L_LAY * H_DIM * H_DIM,      stream);
    launch_cvt(Cm,         wcm,   L_LAY * H_DIM * S_DIM,      stream);
    launch_cvt(out_proj_w, wout,  D_INOUT * H_DIM,            stream);

    // ---------------- h = x @ in_proj_w^T + b ----------------
    launch_gemm(xb, wib, in_proj_b, hbuf, nullptr, NT, H_DIM, D_INOUT, 0, stream);

    // ---------------- layers ----------------
    for (int i = 0; i < L_LAY; ++i) {
        ln_kernel<<<NT, H_DIM, 0, stream>>>(hbuf, ln_g + i * H_DIM, ln_b + i * H_DIM,
                                            xnbuf, xnb);
        // x_state = xn @ ip_w^T + ip_b            [NT, S]
        launch_gemm(xnb, wip + (size_t)i * S_DIM * H_DIM, ip_b + i * S_DIM,
                    xstate, nullptr, NT, S_DIM, H_DIM, 0, stream);
        // gate = sigmoid(xn @ gate_w^T + gate_b)  [NT, H]
        launch_gemm(xnb, wgate + (size_t)i * H_DIM * H_DIM, gate_b + i * H_DIM,
                    gatebf, nullptr, NT, H_DIM, H_DIM, 1, stream);
        // serial recurrence: states[b,t,s] (bf16)
        scan_kernel<<<B_SZ, 256, 0, stream>>>(xstate,
                                              A_mat + (size_t)i * S_DIM * S_DIM,
                                              Bv_vec + i * S_DIM,
                                              bias_A + i * S_DIM, stb);
        // y = states @ Cm^T + bias_C              [NT, H]  (+ bf16 copy)
        launch_gemm(stb, wcm + (size_t)i * H_DIM * S_DIM, bias_C + i * H_DIM,
                    ybuf, yb, NT, H_DIM, S_DIM, 0, stream);
        // yproj = y @ proj_w^T + proj_b           [NT, H]
        launch_gemm(yb, wproj + (size_t)i * H_DIM * H_DIM, proj_b + i * H_DIM,
                    yproj, nullptr, NT, H_DIM, H_DIM, 0, stream);
        // h += gate*yproj + (1-gate)*xn  (also emits h in bf16)
        combine_kernel<<<(NT * H_DIM + 255) / 256, 256, 0, stream>>>(
            hbuf, hb, xnbuf, gatebf, yproj, NT * H_DIM);
    }

    // ---------------- out = h @ out_proj_w^T + b ----------------
    launch_gemm(hb, wout, out_proj_b, (float*)d_out, nullptr,
                NT, D_INOUT, H_DIM, 0, stream);
}
